// LSTM_Autoencoder_69861938037379
// MI455X (gfx1250) — matrix operand order repaired and checked by reading, not compiled, not run on hardware
//
#include <hip/hip_runtime.h>
#include <hip/hip_bf16.h>

// ---------------------------------------------------------------------------
// LSTM autoencoder for MI455X (gfx1250).
//  - All GEMMs use v_wmma_f32_16x16x32_bf16 (f32 accumulate).
//  - Recurrent scans: 8 persistent workgroups, 16 batch rows each, Wr^T
//    bf16-packed in LDS (needs CDNA5's 320KB/WGP LDS). The per-timestep xg
//    slab is double-buffered in LDS via GLOBAL_LOAD_ASYNC_TO_LDS_B128
//    (ASYNCcnt-tracked) so next-step HBM latency overlaps this step's WMMAs.
//  - Gate order i,f,c,o; ALL activations are sigmoid (per reference).
// ---------------------------------------------------------------------------

typedef __bf16 bf16_t;
typedef __attribute__((ext_vector_type(16))) __bf16 v16bf;
typedef __attribute__((ext_vector_type(8)))  float  v8f;
typedef int v4i_async __attribute__((vector_size(16)));  // builtin pointee type

#define AS1 __attribute__((address_space(1)))
#define AS3 __attribute__((address_space(3)))

#if defined(__HIP_DEVICE_COMPILE__) && \
    __has_builtin(__builtin_amdgcn_global_load_async_to_lds_b128) && \
    __has_builtin(__builtin_amdgcn_s_wait_asynccnt)
#define USE_ASYNC 1
#else
#define USE_ASYNC 0
#endif

union FragU { v16bf v; uint4 q[2]; };

#define B_  128
#define T_  512
#define F_  128
#define L_  32
#define U1_ 128
#define U2_ 64

__device__ __forceinline__ float sigmoidf_(float x) {
    return 1.0f / (1.0f + __expf(-x));
}

// A fragment: 16xK(32) bf16 from row-major [M][lda] memory (global or LDS).
// Lanes 0-15: row m0+lane, K = k0+{0..7,16..23}; lanes 16-31: K = +8.
template <typename PT>
__device__ __forceinline__ v16bf load_a_frag(const PT* A, int m0, int k0,
                                             size_t lda, int lane) {
    const int m  = m0 + (lane & 15);
    const int kb = k0 + ((lane >> 4) << 3);
    const bf16_t* p = (const bf16_t*)A + (size_t)m * lda + kb;
    FragU u;
    u.q[0] = *reinterpret_cast<const uint4*>(p);
    u.q[1] = *reinterpret_cast<const uint4*>(p + 16);
    return u.v;
}

// B fragment: K(32)x16 bf16 from PRE-TRANSPOSED weights WT stored [N][K].
// Lanes 0-15: col n0+lane, K = k0+{0..7,16..23}; lanes 16-31: K = +8.
template <typename PT>
__device__ __forceinline__ v16bf load_b_frag(const PT* WT, int n0, int k0,
                                             size_t ldk, int lane) {
    const int n  = n0 + (lane & 15);
    const int kb = k0 + ((lane >> 4) << 3);
    const bf16_t* p = (const bf16_t*)WT + (size_t)n * ldk + kb;
    FragU u;
    u.q[0] = *reinterpret_cast<const uint4*>(p);
    u.q[1] = *reinterpret_cast<const uint4*>(p + 16);
    return u.v;
}

// ---------------------------------------------------------------------------
// Generic bf16 GEMM + bias:  C[M,N] = A[M,K](bf16) @ W[K,N](bf16,T-packed) + b
// block = 256 threads (8 waves); wave w computes tile (blockIdx.x*128 + 16w,
// blockIdx.y*16). K multiple of 32.
// ---------------------------------------------------------------------------
__global__ __launch_bounds__(256)
void gemm_bias_bf16(const bf16_t* __restrict__ A, size_t lda,
                    const bf16_t* __restrict__ WT, int K,
                    const float* __restrict__ bias,
                    float* __restrict__ C, size_t ldc) {
    const int wave = threadIdx.x >> 5;
    const int lane = threadIdx.x & 31;
    const int m0 = blockIdx.x * 128 + wave * 16;
    const int n0 = blockIdx.y * 16;

    v8f acc = {};
    for (int k0 = 0; k0 < K; k0 += 32) {
        v16bf a = load_a_frag(A, m0, k0, lda, lane);
        v16bf b = load_b_frag(WT, n0, k0, (size_t)K, lane);
        acc = __builtin_amdgcn_wmma_f32_16x16x32_bf16(
                false, a, false, b, (short)0, acc, false, false);
    }

    const int n    = n0 + (lane & 15);
    const int mrow = (lane >> 4) << 3;
    const float bv = bias[n];
#pragma unroll
    for (int v = 0; v < 8; ++v) {
        C[(size_t)(m0 + mrow + v) * ldc + n] = acc[v] + bv;
    }
}

// ---------------------------------------------------------------------------
// Persistent LSTM scan. One workgroup owns 16 batch rows for all T steps.
//   z[16,4U] = xg[b, t, :] + h[16,U] @ Wr[U,4U]        (WMMA, Wr^T in LDS)
//   i,f,g,o = sigmoid(z splits);  c = f*c + i*g;  h = o*sigmoid(c)
// REP=true is the RepeatVector decoder input (xg time-invariant: one slab,
// loaded once). Otherwise the xg slab for step t+1 is async-prefetched into
// the alternate LDS buffer while step t computes.
// ---------------------------------------------------------------------------
template <int U, bool REP>
__global__ __launch_bounds__(256)
void lstm_scan_bf16(const float* __restrict__ xg, size_t bstride, size_t tstride,
                    const bf16_t* __restrict__ WrT,   // [4U][U] bf16 packed
                    bf16_t* __restrict__ h_out,       // [B*T][U] bf16
                    int T) {
    constexpr int N4 = 4 * U;
    extern __shared__ __align__(16) char smem[];
    bf16_t* sWrT = (bf16_t*)smem;                       // N4*U bf16
    bf16_t* sH   = sWrT + (size_t)N4 * U;               // 16*U bf16
    float*  sZ   = (float*)(sH + 16 * U);               // 16*N4 f32
    float*  sX   = sZ + 16 * N4;                        // 2 * 16*N4 f32 (async)

    const int tid  = threadIdx.x;
    const int wave = tid >> 5;
    const int lane = tid & 31;
    const int b0   = blockIdx.x * 16;
    constexpr int NTILES = N4 >> 4;
    constexpr int CH     = (16 * N4) / 4;   // b128 chunks per xg slab
    constexpr int PREF   = CH / 256;        // async instrs per wave per slab

    // ---- issue one xg slab (16 rows x N4 f32) into LDS buffer `buf` ----
    auto issue_slab = [&](int t, int buf) {
#if USE_ASYNC
        for (int i = tid; i < CH; i += 256) {
            const int m = i / (N4 / 4);
            const int c = i - m * (N4 / 4);
            const float* g = xg + (size_t)(b0 + m) * bstride +
                             (size_t)t * tstride + (size_t)c * 4;
            float* l = sX + (size_t)buf * 16 * N4 + (size_t)m * N4 + (size_t)c * 4;
            __builtin_amdgcn_global_load_async_to_lds_b128(
                (AS1 v4i_async*)g, (AS3 v4i_async*)l, 0, 0);
        }
#else
        (void)t; (void)buf;
#endif
    };

    // ---- stage Wr^T into LDS; zero h state ----
    {
        const int nvec = (N4 * U) >> 3;  // 8 bf16 per 16B chunk
#if USE_ASYNC
        for (int i = tid; i < nvec; i += 256) {
            __builtin_amdgcn_global_load_async_to_lds_b128(
                (AS1 v4i_async*)(WrT + (size_t)i * 8),
                (AS3 v4i_async*)(sWrT + (size_t)i * 8), 0, 0);
        }
#else
        const uint4* gw = (const uint4*)WrT;
        uint4* sw = (uint4*)sWrT;
        for (int i = tid; i < nvec; i += 256) sw[i] = gw[i];
#endif
        const uint4 zz = make_uint4(0u, 0u, 0u, 0u);
        uint4* sh = (uint4*)sH;
        for (int i = tid; i < (16 * U) >> 3; i += 256) sh[i] = zz;
    }
    float creg[8];
#pragma unroll
    for (int j = 0; j < 8; ++j) creg[j] = 0.0f;

#if USE_ASYNC
    issue_slab(0, 0);
    if (REP) __builtin_amdgcn_s_wait_asynccnt(0);   // single slab: wait now
    // !REP: slab0 completion is guaranteed by the wait inside the first step.
#endif

    const int mrow = (lane >> 4) << 3;

    for (int t = 0; t < T; ++t) {
        const int buf = REP ? 0 : (t & 1);
#if USE_ASYNC
        if (!REP) {
            if (t + 1 < T) {
                issue_slab(t + 1, (t + 1) & 1);
                __builtin_amdgcn_s_wait_asynccnt(PREF); // slab t resident
            } else {
                __builtin_amdgcn_s_wait_asynccnt(0);
            }
        }
#endif
        __syncthreads();  // sH(t-1) + this wave's slab visible workgroup-wide

        // ---- z = xg[t] + h @ Wr  (each wave: NTILES/8 N-tiles) ----
        for (int nt = wave; nt < NTILES; nt += 8) {
            const int n0   = nt << 4;
            const int ncol = n0 + (lane & 15);
            v8f acc;
#if USE_ASYNC
            const float* xrow = sX + (size_t)buf * 16 * N4 + ncol;
#pragma unroll
            for (int v = 0; v < 8; ++v)
                acc[v] = xrow[(size_t)(mrow + v) * N4];
#else
            const float* xp = xg + (size_t)b0 * bstride +
                              (size_t)t * tstride + ncol;
#pragma unroll
            for (int v = 0; v < 8; ++v)
                acc[v] = xp[(size_t)(mrow + v) * bstride];
#endif
#pragma unroll
            for (int k0 = 0; k0 < U; k0 += 32) {
                v16bf a = load_a_frag(sH, 0, k0, (size_t)U, lane);
                v16bf b = load_b_frag(sWrT, n0, k0, (size_t)U, lane);
                acc = __builtin_amdgcn_wmma_f32_16x16x32_bf16(
                        false, a, false, b, (short)0, acc, false, false);
            }
#pragma unroll
            for (int v = 0; v < 8; ++v)
                sZ[(size_t)(mrow + v) * N4 + ncol] = acc[v];
        }
        __syncthreads();

        // ---- gates + state update (c in registers: fixed tid->elem map) ----
        constexpr int TOTAL = 16 * U;
#pragma unroll
        for (int j = 0; j < TOTAL / 256; ++j) {
            const int i = tid + j * 256;
            const int m = i / U;
            const int u = i - m * U;
            const float* zr = sZ + (size_t)m * N4;
            const float gi = sigmoidf_(zr[u]);
            const float gf = sigmoidf_(zr[U + u]);
            const float gg = sigmoidf_(zr[2 * U + u]);
            const float go = sigmoidf_(zr[3 * U + u]);
            const float c  = gf * creg[j] + gi * gg;
            creg[j] = c;
            const float h = go * sigmoidf_(c);
            sH[(size_t)m * U + u] = (bf16_t)h;
            h_out[((size_t)(b0 + m) * T + t) * (size_t)U + u] = (bf16_t)h;
        }
        // next iteration's leading barrier separates sH writes from reads
    }
}

// ---------------------------------------------------------------------------
// fp32 -> bf16 conversion / transposed weight packing
// ---------------------------------------------------------------------------
__global__ void f2bf_kernel(const float* __restrict__ s, bf16_t* __restrict__ d,
                            size_t n) {
    size_t i = (size_t)blockIdx.x * blockDim.x + threadIdx.x;
    size_t stride = (size_t)gridDim.x * blockDim.x;
    for (; i < n; i += stride) d[i] = (bf16_t)s[i];
}

// src [K][N] fp32 row-major  ->  dst [N][K] bf16
__global__ void f2bf_T_kernel(const float* __restrict__ s, bf16_t* __restrict__ d,
                              int K, int N) {
    size_t n = (size_t)K * N;
    size_t i = (size_t)blockIdx.x * blockDim.x + threadIdx.x;
    size_t stride = (size_t)gridDim.x * blockDim.x;
    for (; i < n; i += stride) {
        int k = (int)(i / N);
        int c = (int)(i - (size_t)k * N);
        d[(size_t)c * K + k] = (bf16_t)s[i];
    }
}

// ---------------------------------------------------------------------------
// Host orchestration
// ---------------------------------------------------------------------------
static inline size_t al256(size_t x) { return (x + 255) & ~(size_t)255; }

extern "C" void kernel_launch(void* const* d_in, const int* in_sizes, int n_in,
                              void* d_out, int out_size, void* d_ws, size_t ws_size,
                              hipStream_t stream) {
    (void)in_sizes; (void)n_in; (void)out_size; (void)ws_size;
    const float* x   = (const float*)d_in[0];
    const float* ek1 = (const float*)d_in[1];
    const float* er1 = (const float*)d_in[2];
    const float* eb1 = (const float*)d_in[3];
    const float* ek2 = (const float*)d_in[4];
    const float* er2 = (const float*)d_in[5];
    const float* eb2 = (const float*)d_in[6];
    const float* lw  = (const float*)d_in[7];
    const float* lb  = (const float*)d_in[8];
    const float* dk1 = (const float*)d_in[9];
    const float* dr1 = (const float*)d_in[10];
    const float* db1 = (const float*)d_in[11];
    const float* dk2 = (const float*)d_in[12];
    const float* dr2 = (const float*)d_in[13];
    const float* db2 = (const float*)d_in[14];
    const float* ow  = (const float*)d_in[15];
    const float* ob  = (const float*)d_in[16];
    float* out = (float*)d_out;

    char* ws = (char*)d_ws;
    size_t off = 0;
    auto carve = [&](size_t bytes) { size_t o = off; off = al256(off + bytes); return o; };

    // bf16 transposed weights [N][K]
    bf16_t* ek1T = (bf16_t*)(ws + carve((size_t)512 * 128 * 2));
    bf16_t* er1T = (bf16_t*)(ws + carve((size_t)512 * 128 * 2));
    bf16_t* ek2T = (bf16_t*)(ws + carve((size_t)256 * 128 * 2));
    bf16_t* er2T = (bf16_t*)(ws + carve((size_t)256 * 64 * 2));
    bf16_t* lwT  = (bf16_t*)(ws + carve((size_t)32  * 64 * 2));
    bf16_t* dk1T = (bf16_t*)(ws + carve((size_t)256 * 32 * 2));
    bf16_t* dr1T = (bf16_t*)(ws + carve((size_t)256 * 64 * 2));
    bf16_t* dk2T = (bf16_t*)(ws + carve((size_t)512 * 64 * 2));
    bf16_t* dr2T = (bf16_t*)(ws + carve((size_t)512 * 128 * 2));
    bf16_t* owT  = (bf16_t*)(ws + carve((size_t)128 * 128 * 2));
    // activations (bf16)
    bf16_t* xbf  = (bf16_t*)(ws + carve((size_t)B_ * T_ * F_  * 2));
    bf16_t* h1   = (bf16_t*)(ws + carve((size_t)B_ * T_ * U1_ * 2));
    bf16_t* h2   = (bf16_t*)(ws + carve((size_t)B_ * T_ * U2_ * 2));
    bf16_t* d1   = (bf16_t*)(ws + carve((size_t)B_ * T_ * U2_ * 2));
    bf16_t* d2   = (bf16_t*)(ws + carve((size_t)B_ * T_ * U1_ * 2));
    bf16_t* zbf  = (bf16_t*)(ws + carve((size_t)B_ * L_ * 2));
    float*  zf32 = (float*) (ws + carve((size_t)B_ * L_ * 4));
    float*  xg   = (float*) (ws + carve((size_t)B_ * T_ * 512 * 4)); // shared fp32 proj buffer

    const dim3 blk(256);
    const int MBT = B_ * T_;  // 65536 rows

    // ---- weight packing + input conversion ----
    f2bf_kernel<<<512, blk, 0, stream>>>(x, xbf, (size_t)MBT * F_);
    f2bf_T_kernel<<<256, blk, 0, stream>>>(ek1, ek1T, 128, 512);
    f2bf_T_kernel<<<256, blk, 0, stream>>>(er1, er1T, 128, 512);
    f2bf_T_kernel<<<128, blk, 0, stream>>>(ek2, ek2T, 128, 256);
    f2bf_T_kernel<<<64,  blk, 0, stream>>>(er2, er2T, 64, 256);
    f2bf_T_kernel<<<8,   blk, 0, stream>>>(lw,  lwT,  64, 32);
    f2bf_T_kernel<<<32,  blk, 0, stream>>>(dk1, dk1T, 32, 256);
    f2bf_T_kernel<<<64,  blk, 0, stream>>>(dr1, dr1T, 64, 256);
    f2bf_T_kernel<<<128, blk, 0, stream>>>(dk2, dk2T, 64, 512);
    f2bf_T_kernel<<<256, blk, 0, stream>>>(dr2, dr2T, 128, 512);
    f2bf_T_kernel<<<64,  blk, 0, stream>>>(ow,  owT,  128, 128);

    // LDS: Wr^T + h(bf16) + z(f32) + 2 async xg slabs (always reserved so the
    // dynamic-LDS size matches whichever path the device compile selected).
    const size_t lds_u1 = (size_t)512 * 128 * 2 + 16 * 128 * 2 + 16 * 512 * 4
                        + 2 * (size_t)16 * 512 * 4;  // ~228KB (CDNA5 WGP LDS)
    const size_t lds_u2 = (size_t)256 * 64  * 2 + 16 * 64  * 2 + 16 * 256 * 4
                        + 2 * (size_t)16 * 256 * 4;  // ~82KB

    // ---- encoder LSTM1: F=128 -> U1=128 ----
    gemm_bias_bf16<<<dim3(MBT / 128, 512 / 16), blk, 0, stream>>>(
        xbf, (size_t)F_, ek1T, 128, eb1, xg, (size_t)512);
    lstm_scan_bf16<U1_, false><<<dim3(B_ / 16), blk, lds_u1, stream>>>(
        xg, (size_t)T_ * 512, (size_t)512, er1T, h1, T_);

    // ---- encoder LSTM2: 128 -> U2=64 ----
    gemm_bias_bf16<<<dim3(MBT / 128, 256 / 16), blk, 0, stream>>>(
        h1, (size_t)U1_, ek2T, 128, eb2, xg, (size_t)256);
    lstm_scan_bf16<U2_, false><<<dim3(B_ / 16), blk, lds_u2, stream>>>(
        xg, (size_t)T_ * 256, (size_t)256, er2T, h2, T_);

    // ---- latent dense: h2[:, T-1] @ lw + lb  -> z[128,32] ----
    gemm_bias_bf16<<<dim3(1, 32 / 16), blk, 0, stream>>>(
        h2 + (size_t)(T_ - 1) * U2_, (size_t)T_ * U2_, lwT, 64, lb, zf32, (size_t)L_);
    f2bf_kernel<<<4, blk, 0, stream>>>(zf32, zbf, (size_t)B_ * L_);

    // ---- decoder LSTM1: RepeatVector(z) -> 64. xg time-invariant (REP) ----
    gemm_bias_bf16<<<dim3(1, 256 / 16), blk, 0, stream>>>(
        zbf, (size_t)L_, dk1T, 32, db1, xg, (size_t)256);
    lstm_scan_bf16<U2_, true><<<dim3(B_ / 16), blk, lds_u2, stream>>>(
        xg, (size_t)256, (size_t)0, dr1T, d1, T_);

    // ---- decoder LSTM2: 64 -> 128 ----
    gemm_bias_bf16<<<dim3(MBT / 128, 512 / 16), blk, 0, stream>>>(
        d1, (size_t)U2_, dk2T, 64, db2, xg, (size_t)512);
    lstm_scan_bf16<U1_, false><<<dim3(B_ / 16), blk, lds_u1, stream>>>(
        xg, (size_t)T_ * 512, (size_t)512, dr2T, d2, T_);

    // ---- output dense: d2 @ ow + ob -> [B,T,F] fp32 ----
    gemm_bias_bf16<<<dim3(MBT / 128, 128 / 16), blk, 0, stream>>>(
        d2, (size_t)U1_, owT, 128, ob, out, (size_t)F_);
}
